// NTreeLSTM_2765958938928
// MI455X (gfx1250) — compile-verified, matrix operand order checked by reading, not run on hardware
//
#include <hip/hip_runtime.h>
#include <hip/hip_bf16.h>

typedef __attribute__((ext_vector_type(16))) __bf16 v16bf;
typedef __attribute__((ext_vector_type(8)))  float  v8f;
typedef __attribute__((ext_vector_type(8)))  __bf16 bf16x8;
typedef __attribute__((ext_vector_type(4)))  __bf16 bf16x4;
typedef __attribute__((ext_vector_type(4)))  float  f32x4;

#define HDIM 256
#define XSZ  256
#define TLEVELS 18
#define NNODES ((1 << TLEVELS) - 1)

// CDNA5 hardware transcendental tanh (TRANS pipe, co-executes with WMMA).
__device__ __forceinline__ float fast_tanh(float x) {
#if __has_builtin(__builtin_amdgcn_tanhf)
    return __builtin_amdgcn_tanhf(x);
#else
    float r;
    asm("v_tanh_f32 %0, %1" : "=v"(r) : "v"(x));
    return r;
#endif
}

// sigmoid(x) = 0.5*tanh(x/2) + 0.5  (branch-free, single TRANS op)
__device__ __forceinline__ float fast_sigmoid(float x) {
    return __builtin_fmaf(fast_tanh(0.5f * x), 0.5f, 0.5f);
}

// ---- A fragment (16x32 bf16) read from an LDS-staged bf16 tile.
// ISA layout (16-bit A 16x32): lane L in [0,15]: M=L, elems 0..7 = K 0..7, 8..15 = K 16..23
//                              lane L in [16,31]: M=L-16, elems 0..7 = K 8..15, 8..15 = K 24..31
__device__ __forceinline__ v16bf load_a_frag_lds(const __bf16* As, int rowStride,
                                                 int kc, int lane) {
    const __bf16* p = As + (lane & 15) * rowStride + kc + ((lane >> 4) << 3);
    bf16x8 lo = *(const bf16x8*)p;
    bf16x8 hi = *(const bf16x8*)(p + 16);
    v16bf r;
#pragma unroll
    for (int i = 0; i < 8; ++i) { r[i] = lo[i]; r[8 + i] = hi[i]; }
    return r;
}

// ---- B fragment: 32x16 bf16 from pre-transposed (column-major-as-rows) bf16 weights.
__device__ __forceinline__ v16bf load_b_frag_bf16(const __bf16* __restrict__ colbase,
                                                  int kc, int lane) {
    const __bf16* p = colbase + kc + ((lane >> 4) << 3);
    bf16x8 lo = *(const bf16x8*)p;
    bf16x8 hi = *(const bf16x8*)(p + 16);
    v16bf r;
#pragma unroll
    for (int i = 0; i < 8; ++i) { r[i] = lo[i]; r[8 + i] = hi[i]; }
    return r;
}

// dst[n*K + k] = (bf16) src[k*Ncols + n]   (transpose + downconvert, one-time prep)
__global__ void transpose_bf16_kernel(const float* __restrict__ src, __bf16* __restrict__ dst,
                                      int K, int Ncols) {
    int idx = blockIdx.x * blockDim.x + threadIdx.x;
    if (idx >= K * Ncols) return;
    int n = idx / K;
    int k = idx - n * K;
    dst[idx] = (__bf16)src[(size_t)k * Ncols + n];
}

// init = tanh(x @ W_w + W_b); cols [0,256)->h, [256,512)->c
// Block: 256 threads = 8 waves; block owns a 16-row tile; wave w does N-tiles w*4..w*4+3.
__global__ __launch_bounds__(256) void init_kernel(const float* __restrict__ x,
                                                   const __bf16* __restrict__ WT,
                                                   const float* __restrict__ Wb,
                                                   float* __restrict__ h,
                                                   float* __restrict__ c) {
    __shared__ __bf16 As[16 * XSZ];           // 8 KB staged A tile
    int tid = threadIdx.x, wid = tid >> 5, lane = tid & 31;
    int m0 = blockIdx.x * 16;

    // Stage 16x256 fp32 -> bf16 LDS, zero-fill out-of-range rows (no masking in hot loop)
    for (int i = tid; i < 16 * XSZ / 4; i += 256) {
        int row = i >> 6;                     // 64 float4 groups per row
        int kq  = (i & 63) << 2;
        int m   = m0 + row;
        bf16x4 v4;
        if (m < NNODES) {
            f32x4 v = *(const f32x4*)(x + (size_t)m * XSZ + kq);
            v4[0] = (__bf16)v.x; v4[1] = (__bf16)v.y;
            v4[2] = (__bf16)v.z; v4[3] = (__bf16)v.w;
        } else {
            v4[0] = v4[1] = v4[2] = v4[3] = (__bf16)0.0f;
        }
        *(bf16x4*)(&As[row * XSZ + kq]) = v4;
    }
    __syncthreads();

    const __bf16* bcol[4];
    float bias[4];
#pragma unroll
    for (int t = 0; t < 4; ++t) {
        int col = (wid * 4 + t) * 16 + (lane & 15);
        bcol[t] = WT + (size_t)col * XSZ;
        bias[t] = Wb[col];
    }

    v8f acc[4];
#pragma unroll
    for (int t = 0; t < 4; ++t) acc[t] = (v8f){};

#pragma unroll 2
    for (int kc = 0; kc < XSZ; kc += 32) {
        v16bf a = load_a_frag_lds(As, XSZ, kc, lane);
#pragma unroll
        for (int t = 0; t < 4; ++t) {
            v16bf b = load_b_frag_bf16(bcol[t], kc, lane);
            acc[t] = __builtin_amdgcn_wmma_f32_16x16x32_bf16(false, a, false, b,
                                                             (short)0, acc[t], false, false);
        }
    }

#pragma unroll
    for (int t = 0; t < 4; ++t) {
        int col  = (wid * 4 + t) * 16 + (lane & 15);
        int colH = col & 255;
        bool isC = col >= HDIM;
#pragma unroll
        for (int r = 0; r < 8; ++r) {
            int row = m0 + r + ((lane >> 4) << 3);
            if (row < NNODES) {
                float v = fast_tanh(acc[t][r] + bias[t]);
                if (isC) c[(size_t)row * HDIM + colH] = v;
                else     h[(size_t)row * HDIM + colH] = v;
            }
        }
    }
}

// One tree level: gates = act(ch_h @ [Uf|Uiou] + b); c_new = i*u + f0*c0 + f1*c1; h_new = o*tanh(c_new)
// Block: 320 threads = 10 waves; block owns 16 nodes; wave w does N-tiles w*8..w*8+7 (80 total).
__global__ __launch_bounds__(320) void level_kernel(float* __restrict__ h,
                                                    float* __restrict__ c,
                                                    const __bf16* __restrict__ UfT,
                                                    const __bf16* __restrict__ UiouT,
                                                    const float* __restrict__ Ufb,
                                                    const float* __restrict__ Uioub,
                                                    int cnt, int off, int coff) {
    __shared__ __bf16 As[16 * 512];           // 16 KB staged children-h tile (bf16)
    __shared__ __bf16 G[16 * 1280];           // 40 KB activated gates
    int tid = threadIdx.x, wid = tid >> 5, lane = tid & 31;
    int m0 = blockIdx.x * 16;

    const float* A = h + (size_t)coff * HDIM; // (cnt x 512) row-major view of children h

    // Stage 16x512 fp32 -> bf16 LDS, zero-fill out-of-range rows
    for (int i = tid; i < 16 * 512 / 4; i += 320) {
        int row = i >> 7;                     // 128 float4 groups per row
        int kq  = (i & 127) << 2;
        int m   = m0 + row;
        bf16x4 v4;
        if (m < cnt) {
            f32x4 v = *(const f32x4*)(A + (size_t)m * 512 + kq);
            v4[0] = (__bf16)v.x; v4[1] = (__bf16)v.y;
            v4[2] = (__bf16)v.z; v4[3] = (__bf16)v.w;
        } else {
            v4[0] = v4[1] = v4[2] = v4[3] = (__bf16)0.0f;
        }
        *(bf16x4*)(&As[row * 512 + kq]) = v4;
    }
    __syncthreads();

    const __bf16* bcol[8];
    float bias[8];
#pragma unroll
    for (int t = 0; t < 8; ++t) {
        int col = (wid * 8 + t) * 16 + (lane & 15);
        bcol[t] = (col < 512) ? (UfT + (size_t)col * 512)
                              : (UiouT + (size_t)(col - 512) * 512);
        bias[t] = (col < 512) ? Ufb[col] : Uioub[col - 512];
    }

    v8f acc[8];
#pragma unroll
    for (int t = 0; t < 8; ++t) acc[t] = (v8f){};

#pragma unroll 2
    for (int kc = 0; kc < 512; kc += 32) {
        v16bf a = load_a_frag_lds(As, 512, kc, lane);
#pragma unroll
        for (int t = 0; t < 8; ++t) {
            v16bf b = load_b_frag_bf16(bcol[t], kc, lane);
            acc[t] = __builtin_amdgcn_wmma_f32_16x16x32_bf16(false, a, false, b,
                                                             (short)0, acc[t], false, false);
        }
    }

#pragma unroll
    for (int t = 0; t < 8; ++t) {
        int col  = (wid * 8 + t) * 16 + (lane & 15);
        bool isU = col >= 1024;               // u gets tanh; f,i,o get sigmoid
#pragma unroll
        for (int r = 0; r < 8; ++r) {
            int ml  = r + ((lane >> 4) << 3);
            float g = acc[t][r] + bias[t];
            G[ml * 1280 + col] = (__bf16)(isU ? fast_tanh(g) : fast_sigmoid(g));
        }
    }
    __syncthreads();

    for (int idx = tid; idx < 16 * HDIM; idx += 320) {
        int ml = idx >> 8;
        int j  = idx & 255;
        int m  = m0 + ml;
        if (m < cnt) {
            float f0 = (float)G[ml * 1280 + j];
            float f1 = (float)G[ml * 1280 + 256 + j];
            float ig = (float)G[ml * 1280 + 512 + j];
            float og = (float)G[ml * 1280 + 768 + j];
            float ug = (float)G[ml * 1280 + 1024 + j];
            float c0 = c[(size_t)(coff + 2 * m)     * HDIM + j];
            float c1 = c[(size_t)(coff + 2 * m + 1) * HDIM + j];
            float cn = ig * ug + f0 * c0 + f1 * c1;
            float hn = og * fast_tanh(cn);
            c[(size_t)(off + m) * HDIM + j] = cn;
            h[(size_t)(off + m) * HDIM + j] = hn;
        }
    }
}

extern "C" void kernel_launch(void* const* d_in, const int* in_sizes, int n_in,
                              void* d_out, int out_size, void* d_ws, size_t ws_size,
                              hipStream_t stream) {
    const float* x      = (const float*)d_in[0];
    const float* W_w    = (const float*)d_in[1];
    const float* W_b    = (const float*)d_in[2];
    const float* Uiou_w = (const float*)d_in[3];
    const float* Uiou_b = (const float*)d_in[4];
    const float* Uf_w   = (const float*)d_in[5];
    const float* Uf_b   = (const float*)d_in[6];

    float* h = (float*)d_out;

    char* ws = (char*)d_ws;
    float*  c      = (float*)ws;                                // N*256 fp32
    size_t  cbytes = (size_t)NNODES * HDIM * sizeof(float);
    __bf16* WT    = (__bf16*)(ws + cbytes);                     // (512 x 256)
    __bf16* UfT   = WT  + (size_t)512 * 256;                    // (512 x 512)
    __bf16* UiouT = UfT + (size_t)512 * 512;                    // (768 x 512)

    // Weight transpose + bf16 downconvert (one-time, tiny)
    transpose_bf16_kernel<<<(512 * 256 + 255) / 256, 256, 0, stream>>>(W_w,    WT,    256, 512);
    transpose_bf16_kernel<<<(512 * 512 + 255) / 256, 256, 0, stream>>>(Uf_w,   UfT,   512, 512);
    transpose_bf16_kernel<<<(768 * 512 + 255) / 256, 256, 0, stream>>>(Uiou_w, UiouT, 512, 768);

    // Leaf/init GEMM: one block per 16-row tile, 8 waves cover the 32 N-tiles (4 each)
    init_kernel<<<(NNODES + 15) / 16, 256, 0, stream>>>(x, WT, W_b, h, c);

    // Bottom-up levels
    for (int l = TLEVELS - 2; l >= 0; --l) {
        int cnt  = 1 << l;
        int off  = cnt - 1;
        int coff = 2 * cnt - 1;
        int blocks = (cnt + 15) / 16;
        level_kernel<<<blocks, 320, 0, stream>>>(h, c, UfT, UiouT, Uf_b, Uiou_b,
                                                 cnt, off, coff);
    }
}
